// RowLSTM_7765300871324
// MI455X (gfx1250) — matrix operand order repaired and checked by reading, not compiled, not run on hardware
//
#include <hip/hip_runtime.h>
#include <hip/hip_bf16.h>
#include <math.h>
#include <stdint.h>

// Flip to 0 if the async-to-LDS mnemonic is rejected by the assembler.
#ifndef USE_ASYNC_LDS
#define USE_ASYNC_LDS 1
#endif

// ---------------- problem constants ----------------
#define B_    32
#define CIN   128
#define COUT  128
#define H_    64
#define W_    64
#define FOURC 512                 // 4*COUT
#define KIN   256                 // 2 taps * CIN (tap 2 masked to zero)
#define KSS   384                 // 3 taps * COUT
#define NIN   (B_*H_*W_)          // 131072 columns for input GEMM
#define NSS   (B_*W_)             // 2048 columns for state GEMM

// ---------------- workspace layout (bytes) ----------------
#define XMAT_OFF   0ull
#define GATES_OFF  (XMAT_OFF + (unsigned long long)NIN * KIN * 2ull)
#define WIN_OFF    (GATES_OFF + (unsigned long long)H_ * B_ * FOURC * W_ * 4ull)
#define WSS_OFF    (WIN_OFF + (unsigned long long)FOURC * KIN * 2ull)
#define HMAT0_OFF  (WSS_OFF + (unsigned long long)FOURC * KSS * 2ull)
#define HMAT1_OFF  (HMAT0_OFF + (unsigned long long)NSS * KSS * 2ull)
#define CST_OFF    (HMAT1_OFF + (unsigned long long)NSS * KSS * 2ull)

typedef __attribute__((ext_vector_type(16))) __bf16 v16bf;
typedef __attribute__((ext_vector_type(8)))  __bf16 v8bf;
typedef __attribute__((ext_vector_type(8)))  float  v8f;

// ---------------- WMMA fragment loaders (wave32, 16x16x32 bf16) ----------------
// A (16x32, MxK), [M][K] row-major.  Lane l: m = l&15, kh = l>>4.
// Elements 0..7 -> K = k0 + kh*8 + e; elements 8..15 -> K = k0 + 16 + kh*8 + (e-8).
__device__ __forceinline__ v16bf load_a_frag(const __bf16* __restrict__ A, int K,
                                             int row, int k0, int kh) {
  const __bf16* p = A + (size_t)row * K + k0 + kh * 8;
  v8bf lo = *(const v8bf*)p;
  v8bf hi = *(const v8bf*)(p + 16);
  v16bf r;
#pragma unroll
  for (int i = 0; i < 8; ++i) { r[i] = lo[i]; r[8 + i] = hi[i]; }
  return r;
}

// B (32x16, KxN), stored [N][K] row-major.  Lane l: n = l&15, kh = l>>4.
// Elements 0..7 -> K = k0 + kh*8 + e; elements 8..15 -> K = k0 + 16 + kh*8 + (e-8).
__device__ __forceinline__ v16bf load_b_frag(const __bf16* __restrict__ Bm, int K,
                                             int col, int k0, int kh) {
  const __bf16* p = Bm + (size_t)col * K + k0 + kh * 8;
  v8bf lo = *(const v8bf*)p;
  v8bf hi = *(const v8bf*)(p + 16);
  v16bf r;
#pragma unroll
  for (int i = 0; i < 8; ++i) { r[i] = lo[i]; r[8 + i] = hi[i]; }
  return r;
}

// A-fragment from LDS (same layout, K = KSS).
__device__ __forceinline__ v16bf load_a_frag_lds(const __bf16* L, int row, int k0, int kh) {
  const __bf16* p = L + row * KSS + k0 + kh * 8;
  v8bf lo = *(const v8bf*)p;
  v8bf hi = *(const v8bf*)(p + 16);
  v16bf r;
#pragma unroll
  for (int i = 0; i < 8; ++i) { r[i] = lo[i]; r[8 + i] = hi[i]; }
  return r;
}

// ---------------- pack kernels ----------------
__global__ __launch_bounds__(256) void pack_weights_kernel(
    const float* __restrict__ w_in, const float* __restrict__ w_ss,
    __bf16* __restrict__ Win, __bf16* __restrict__ Wss) {
  int tid = blockIdx.x * 256 + threadIdx.x;       // over FOURC*KSS = 196608
  if (tid >= FOURC * KSS) return;
  int m  = tid / KSS;
  int r  = tid % KSS;
  int t  = r / COUT;
  int ci = r % COUT;
  Wss[(size_t)m * KSS + r] = (__bf16)w_ss[((size_t)m * COUT + ci) * 3 + t];
  if (t < 2)
    Win[(size_t)m * KIN + t * CIN + ci] = (__bf16)w_in[((size_t)m * CIN + ci) * 3 + t];
}

__global__ __launch_bounds__(256) void init_state_kernel(
    const float* __restrict__ h0, const float* __restrict__ c0,
    __bf16* __restrict__ Hm0, __bf16* __restrict__ Hm1, float* __restrict__ cst) {
  int tid = blockIdx.x * 256 + threadIdx.x;       // over B*COUT*W = 262144
  int b = tid / (COUT * W_);
  int c = (tid / W_) % COUT;
  int w = tid % W_;
  cst[((size_t)b * COUT + c) * W_ + w] = c0[c * W_ + w];
#pragma unroll
  for (int t = 0; t < 3; ++t) {
    int ws = w + t - 1;
    __bf16 v = (ws >= 0 && ws < W_) ? (__bf16)h0[c * W_ + ws] : (__bf16)0.0f;
    size_t idx = ((size_t)b * W_ + w) * KSS + t * COUT + c;
    Hm0[idx] = v;
    Hm1[idx] = (__bf16)0.0f;   // boundary zeros persist; interior overwritten by row 0
  }
}

__global__ __launch_bounds__(256) void pack_x_kernel(
    const float* __restrict__ x, __bf16* __restrict__ Xm) {
  int tid = blockIdx.x * 256 + threadIdx.x;       // over NIN*CIN = 16,777,216
  int n  = tid / CIN;
  int ci = tid % CIN;
  int b = n / (H_ * W_);
  int h = (n / W_) % H_;
  int w = n % W_;
#pragma unroll
  for (int t = 0; t < 2; ++t) {
    int ws = w + t - 1;
    __bf16 v = (ws >= 0) ? (__bf16)x[(((size_t)b * CIN + ci) * H_ + h) * W_ + ws]
                         : (__bf16)0.0f;
    Xm[(size_t)n * KIN + t * CIN + ci] = v;
  }
}

// ---------------- input conv GEMM: gates[h][b][m][w] = b_in[m] + Win @ Xmat ----------------
__global__ __launch_bounds__(256) void gemm_in_kernel(
    const __bf16* __restrict__ Wm, const __bf16* __restrict__ Xm,
    const float* __restrict__ b_in, float* __restrict__ gates) {
  int lane = threadIdx.x & 31;
  int wave = threadIdx.x >> 5;
  int gw   = blockIdx.x * 8 + wave;               // 262144 waves total
  int mt   = gw & 31;
  int nt   = gw >> 5;
  int kh   = lane >> 4;
  int lm   = lane & 15;
  int a_row = mt * 16 + lm;
  int n_col = nt * 16 + lm;

  v8f acc;
#pragma unroll
  for (int v = 0; v < 8; ++v) acc[v] = b_in[mt * 16 + v + kh * 8];

#pragma unroll
  for (int kk = 0; kk < KIN; kk += 32) {
    v16bf a = load_a_frag(Wm, KIN, a_row, kk, kh);
    v16bf b = load_b_frag(Xm, KIN, n_col, kk, kh);
    acc = __builtin_amdgcn_wmma_f32_16x16x32_bf16(false, a, false, b,
                                                  (short)0, acc, false, false);
  }

  int b = n_col / (H_ * W_);
  int h = (n_col / W_) % H_;
  int w = n_col % W_;
#pragma unroll
  for (int v = 0; v < 8; ++v) {
    int m = mt * 16 + v + kh * 8;
    gates[(((size_t)h * B_ + b) * FOURC + m) * W_ + w] = acc[v];
  }
}

// ---------------- fused recurrent row step ----------------
// One block = one channel-tile ct (8 blocks of the grid's fast dim).  All 8 waves share
// the Wss slice for the 4 gate groups of ct (64 rows x 384 K bf16 = 48 KB), staged to LDS
// with async global->LDS copies.  Each wave computes 4 accumulator tiles (one per gate
// group) over the same B columns, then applies the LSTM pointwise in-register.
__global__ __launch_bounds__(256) void row_step_kernel(
    const __bf16* __restrict__ Wss, const __bf16* __restrict__ HmIn,
    __bf16* __restrict__ HmOut, const float* __restrict__ gates,
    const float* __restrict__ b_ss, const float* __restrict__ x,
    float* __restrict__ cst, float* __restrict__ out, int h) {
  __shared__ __bf16 smem[64 * KSS];               // 49152 B

  const int tid  = threadIdx.x;
  const int lane = tid & 31;
  const int wave = tid >> 5;
  const int ct   = blockIdx.x & 7;                // 8 channel tiles
  const int ntg  = blockIdx.x >> 3;               // 16 nt groups
  const int nt   = ntg * 8 + wave;                // 128 N-tiles
  const int kh   = lane >> 4;
  const int lm   = lane & 15;
  const int n_col = nt * 16 + lm;
  const int b    = n_col >> 6;                    // n = b*W + w
  const int w    = n_col & 63;

  // ---- stage Wss slice (rows g*128 + ct*16 + [0,16), g = 0..3) into LDS ----
#pragma unroll
  for (int it = 0; it < 12; ++it) {
    int chunk = it * 256 + tid;                   // 0..3071, 16 B each
    int boff  = chunk * 16;
    int r     = boff / (KSS * 2);                 // LDS row 0..63
    int o     = boff % (KSS * 2);                 // byte within row
    int g     = r >> 4;
    int mrow  = g * 128 + ct * 16 + (r & 15);
    const char* gp = (const char*)Wss + (size_t)mrow * (KSS * 2) + o;
#if USE_ASYNC_LDS
    unsigned lds_off = (unsigned)(uintptr_t)((char*)smem + boff);
    asm volatile("global_load_async_to_lds_b128 %0, %1, off"
                 :: "v"(lds_off), "v"((unsigned long long)(uintptr_t)gp)
                 : "memory");
#else
    *(v8bf*)((char*)smem + boff) = *(const v8bf*)gp;
#endif
  }
#if USE_ASYNC_LDS
  asm volatile("s_wait_asynccnt 0x0" ::: "memory");
#endif
  __syncthreads();

  // ---- accumulators: gates tile + recurrent bias, one per gate group ----
  v8f acc[4];
#pragma unroll
  for (int g = 0; g < 4; ++g) {
#pragma unroll
    for (int v = 0; v < 8; ++v) {
      int m = g * 128 + ct * 16 + v + kh * 8;
      acc[g][v] = gates[(((size_t)h * B_ + b) * FOURC + m) * W_ + w] + b_ss[m];
    }
  }

  // ---- K loop: B fragment shared by the 4 gate-group WMMAs ----
#pragma unroll
  for (int kk = 0; kk < KSS; kk += 32) {          // 12 iterations -> 48 WMMA
    v16bf bf = load_b_frag(HmIn, KSS, n_col, kk, kh);
#pragma unroll
    for (int g = 0; g < 4; ++g) {
      v16bf af = load_a_frag_lds(smem, g * 16 + lm, kk, kh);
      acc[g] = __builtin_amdgcn_wmma_f32_16x16x32_bf16(false, af, false, bf,
                                                       (short)0, acc[g], false, false);
    }
  }

  // ---- LSTM pointwise, state update, output, next-row B-matrix scatter ----
#pragma unroll
  for (int v = 0; v < 8; ++v) {
    int c = ct * 16 + v + kh * 8;                 // channel 0..127
    float og = 1.0f / (1.0f + __expf(-acc[0][v]));
    float fg = 1.0f / (1.0f + __expf(-acc[1][v]));
    float ig = 1.0f / (1.0f + __expf(-acc[2][v]));
    float cg = tanhf(acc[3][v]);

    size_t cidx = ((size_t)b * COUT + c) * W_ + w;
    float cp = cst[cidx];
    float cn = fg * cp + ig * cg;
    float hn = og * tanhf(cn);
    cst[cidx] = cn;

    out[(((size_t)b * COUT + c) * H_ + h) * W_ + w] =
        hn + x[(((size_t)b * CIN + c) * H_ + h) * W_ + w];

    __bf16 hb = (__bf16)hn;
    if (w + 1 < W_) HmOut[((size_t)b * W_ + (w + 1)) * KSS + 0 * COUT + c] = hb;  // tap 0
    HmOut[((size_t)b * W_ + w) * KSS + 1 * COUT + c] = hb;                        // tap 1
    if (w >= 1)     HmOut[((size_t)b * W_ + (w - 1)) * KSS + 2 * COUT + c] = hb;  // tap 2
  }
}

// ---------------- launch ----------------
extern "C" void kernel_launch(void* const* d_in, const int* in_sizes, int n_in,
                              void* d_out, int out_size, void* d_ws, size_t ws_size,
                              hipStream_t stream) {
  const float* x    = (const float*)d_in[0];
  const float* w_in = (const float*)d_in[1];
  const float* b_in = (const float*)d_in[2];
  const float* w_ss = (const float*)d_in[3];
  const float* b_ss = (const float*)d_in[4];
  const float* h0   = (const float*)d_in[5];
  const float* c0   = (const float*)d_in[6];
  float* out = (float*)d_out;

  char* ws = (char*)d_ws;
  __bf16* Xm    = (__bf16*)(ws + XMAT_OFF);
  float*  gates = (float*)(ws + GATES_OFF);
  __bf16* Win   = (__bf16*)(ws + WIN_OFF);
  __bf16* Wss   = (__bf16*)(ws + WSS_OFF);
  __bf16* Hm[2] = { (__bf16*)(ws + HMAT0_OFF), (__bf16*)(ws + HMAT1_OFF) };
  float*  cst   = (float*)(ws + CST_OFF);

  pack_weights_kernel<<<(FOURC * KSS + 255) / 256, 256, 0, stream>>>(w_in, w_ss, Win, Wss);
  init_state_kernel<<<(B_ * COUT * W_) / 256, 256, 0, stream>>>(h0, c0, Hm[0], Hm[1], cst);
  pack_x_kernel<<<(NIN * CIN) / 256, 256, 0, stream>>>(x, Xm);

  // Input conv GEMM: 32 M-tiles x 8192 N-tiles, 8 waves / block.
  gemm_in_kernel<<<(32 * 8192) / 8, 256, 0, stream>>>(Win, Xm, b_in, gates);

  // Fused recurrent scan: one kernel per row, double-buffered Hmat.
  for (int h = 0; h < H_; ++h) {
    row_step_kernel<<<128, 256, 0, stream>>>(Wss, Hm[h & 1], Hm[(h + 1) & 1],
                                             gates, b_ss, x, cst, out, h);
  }
}
